// CrossStreamAttention_47141561040923
// MI455X (gfx1250) — matrix operand order repaired and checked by reading, not compiled, not run on hardware
//
#include <hip/hip_runtime.h>
#include <hip/hip_bf16.h>

typedef _Float16 hvec4  __attribute__((ext_vector_type(4)));
typedef _Float16 hvec8  __attribute__((ext_vector_type(8)));
typedef _Float16 hvec16 __attribute__((ext_vector_type(16)));
typedef float    fvec8  __attribute__((ext_vector_type(8)));
typedef unsigned int v4u __attribute__((ext_vector_type(4)));
typedef int v4i __attribute__((ext_vector_type(4)));
typedef int v8i __attribute__((ext_vector_type(8)));

#define B_TOT   32768
#define DMODEL  512
#define NHEAD   8
#define DHEAD   64
#define D3      1536
#define D2      1024

union HU { hvec16 v; hvec8 p[2]; };

// A-fragment (16x32 f16, MxK): lane (m=lane&15, hf=lane>>4), element e -> K = kbase + 8*hf + (e&7) + 16*(e>>3)
static __device__ __forceinline__ hvec16 ldA(const _Float16* base, int stride, int m, int kbase, int hf) {
  HU u;
  const _Float16* p = base + m * stride + kbase + 8 * hf;
  u.p[0] = *(const hvec8*)(p);
  u.p[1] = *(const hvec8*)(p + 16);
  return u.v;
}
// B-fragment (32x16 f16, KxN): lane (n=lane&15, hf=lane>>4), element e -> K = kbase + 8*hf + (e&7) + 16*(e>>3)
static __device__ __forceinline__ hvec16 ldB(const _Float16* base, int stride, int n, int kbase, int hf) {
  HU u;
  const _Float16* p = base + n * stride + kbase + 8 * hf;
  u.p[0] = *(const hvec8*)(p);
  u.p[1] = *(const hvec8*)(p + 16);
  return u.v;
}
static __device__ __forceinline__ fvec8 wmma16(hvec16 a, hvec16 b, fvec8 c) {
  return __builtin_amdgcn_wmma_f32_16x16x32_f16(false, a, false, b, (short)0, c, false, false);
}
static __device__ __forceinline__ const float* pick3(const float* a, const float* b, const float* c, int s) {
  return s == 0 ? a : (s == 1 ? b : c);
}
static __device__ __forceinline__ unsigned ldsoff(const void* p) {
  // flat LDS address: low 32 bits are the LDS byte offset (ISA 10.2 aperture mapping)
  return (unsigned)(size_t)p;
}

// TDM 2-D tile load: rows x cols_halves f16 tile, global row stride = row_stride_elems,
// LDS padded by pad_amount dwords every 2^(pad_interval+1) dwords (reproduces padded row stride).
static __device__ __forceinline__ void tdm_load_2d(
    unsigned lds_byte_off, const _Float16* gaddr,
    unsigned rows, unsigned cols_halves, unsigned row_stride_elems,
    unsigned pad_interval_code, unsigned pad_amount_code)
{
  unsigned long long ga = (unsigned long long)(size_t)gaddr;
  v4u g0;
  g0[0] = 1u;                                           // count=1, user descriptor
  g0[1] = lds_byte_off;                                 // lds_addr
  g0[2] = (unsigned)ga;                                 // global_addr[31:0]
  g0[3] = (unsigned)((ga >> 32) & 0x1FFFFFFu) | (2u << 30);  // global_addr[56:32] | type=2
  v8i g1;
  g1[0] = (int)((1u << 16) | (1u << 20) |               // data_size=2B, pad_enable
                (pad_interval_code << 22) | (pad_amount_code << 25));
  g1[1] = (int)((cols_halves & 0xFFFFu) << 16);         // tensor_dim0[15:0]
  g1[2] = (int)((cols_halves >> 16) | ((rows & 0xFFFFu) << 16));  // tensor_dim0[31:16] | tensor_dim1[15:0]
  g1[3] = (int)((rows >> 16) | (cols_halves << 16));    // tensor_dim1[31:16] | tile_dim0
  g1[4] = (int)rows;                                    // tile_dim1 (tile_dim2=0)
  g1[5] = (int)row_stride_elems;                        // tensor_dim0_stride[31:0]
  g1[6] = 0;                                            // stride0[47:32] | stride1[15:0]
  g1[7] = 0;
  v4i z4 = {0, 0, 0, 0};
#if defined(__clang_major__) && (__clang_major__ >= 23)
  v8i z8 = {0, 0, 0, 0, 0, 0, 0, 0};
  __builtin_amdgcn_tensor_load_to_lds(g0, g1, z4, z4, z8, 0);
#else
  __builtin_amdgcn_tensor_load_to_lds(g0, g1, z4, z4, 0);
#endif
}

// ---------------------------------------------------------------- K0: fp32 -> f16 weight convert
__global__ void k0_f32_to_f16(const float* __restrict__ src, _Float16* __restrict__ dst, int n) {
  int i = blockIdx.x * blockDim.x + threadIdx.x;
  if (i < n) dst[i] = (_Float16)src[i];
}

// ---------------------------------------------------------------- K1: QKV + attention + out-proj + gate -> gated[B,1536] f16
// 768 threads = 24 waves; 32 batch rows / WG; M = 96 (row m = s*32 + t).
__global__ __launch_bounds__(768) void k1_attn_gate(
    const float* __restrict__ xk, const float* __restrict__ xc, const float* __restrict__ xs,
    const _Float16* __restrict__ winh, const float* __restrict__ b_in,
    const _Float16* __restrict__ wouth, const float* __restrict__ b_out,
    const float* __restrict__ w_gate, const float* __restrict__ b_gate,
    _Float16* __restrict__ gated)
{
  __shared__ __align__(16) unsigned char smem[57216];
  _Float16* S    = (_Float16*)smem;                        // aliased 41472B region
  _Float16* Ast  = S;                                      // [96][72]  (QKV gemm A chunk)
  _Float16* Bst  = S + 96 * 72;                            // [192][72] (QKV gemm B chunk)
  _Float16* qkv  = S;                                      // [96][200] (q|k|v of one head)
  _Float16* Wst  = S;                                      // [512][40] (W_out K-chunk)
  _Float16* ost  = (_Float16*)(smem + 41472);              // [96][72]  (per-head attn out)
  float*    attw = (float*)(smem + 41472 + 13824);         // [32][9]   scores/attn
  float*    lgts = (float*)(smem + 41472 + 13824 + 1152);  // [32][3]   gate logits
  float*    gts  = lgts + 96;                              // [32][3]   gate weights

  const int tid  = threadIdx.x;
  const int wave = tid >> 5;
  const int lane = tid & 31;
  const int ln   = lane & 15;
  const int hf   = lane >> 4;
  const int mt   = wave % 6;
  const int ngp  = wave / 6;                 // 0..3
  const int b0   = blockIdx.x * 32;

  fvec8 Cop[8];                              // out-proj accum: [mt tile][N = ngp*128 .. +128)
  #pragma unroll
  for (int j = 0; j < 8; ++j)
    #pragma unroll
    for (int r = 0; r < 8; ++r) Cop[j][r] = 0.f;

  for (int h = 0; h < NHEAD; ++h) {
    fvec8 Cq[3];
    #pragma unroll
    for (int j = 0; j < 3; ++j)
      #pragma unroll
      for (int r = 0; r < 8; ++r) Cq[j][r] = 0.f;

    for (int kc = 0; kc < DMODEL; kc += 64) {
      __syncthreads();
      // TDM: stage 3x (64 rows x 64 halves) q/k/v blocks of w_in for head h (padded 72-half rows)
      if (wave == 0) {
        #pragma unroll
        for (int blk = 0; blk < 3; ++blk)
          tdm_load_2d(ldsoff(Bst + blk * 64 * 72),
                      winh + ((size_t)(blk * DMODEL + h * DHEAD)) * DMODEL + kc,
                      64, 64, DMODEL, 4, 3);
      }
      // stage A (convert fp32->f16), float4 -> hvec4, overlaps with TDM
      for (int q = tid; q < 96 * 16; q += 768) {
        int m = q >> 4, kq = (q & 15) * 4;
        const float* xp = pick3(xk, xc, xs, m >> 5);
        const float4 v = *(const float4*)&xp[(size_t)(b0 + (m & 31)) * DMODEL + kc + kq];
        hvec4 o;
        o[0] = (_Float16)v.x; o[1] = (_Float16)v.y; o[2] = (_Float16)v.z; o[3] = (_Float16)v.w;
        *(hvec4*)&Ast[m * 72 + kq] = o;
      }
      if (wave == 0) __builtin_amdgcn_s_wait_tensorcnt(0);
      __syncthreads();
      #pragma unroll
      for (int ks = 0; ks < 2; ++ks) {
        hvec16 a = ldA(Ast, 72, mt * 16 + ln, ks * 32, hf);
        #pragma unroll
        for (int j = 0; j < 3; ++j) {
          hvec16 b = ldB(Bst, 72, (ngp * 3 + j) * 16 + ln, ks * 32, hf);
          Cq[j] = wmma16(a, b, Cq[j]);
        }
      }
    }
    __syncthreads();
    // ---- write qkv (+b_in) as f16 into LDS (cols: 0..63 q, 64..127 k, 128..191 v)
    #pragma unroll
    for (int j = 0; j < 3; ++j) {
      int col = (ngp * 3 + j) * 16 + ln;
      float bi = b_in[(col >> 6) * DMODEL + h * DHEAD + (col & 63)];
      #pragma unroll
      for (int r = 0; r < 8; ++r) {
        int row = mt * 16 + r + 8 * hf;
        qkv[row * 200 + col] = (_Float16)(Cq[j][r] + bi);
      }
    }
    __syncthreads();
    // ---- attention scores (S=3): 288 dots of length 64
    if (tid < 288) {
      int t = tid / 9, rem = tid % 9;
      int sq = rem / 3, sk = rem % 3;
      const _Float16* qr = &qkv[(sq * 32 + t) * 200];
      const _Float16* kr = &qkv[(sk * 32 + t) * 200 + 64];
      float acc = 0.f;
      for (int d = 0; d < 64; ++d) acc += (float)qr[d] * (float)kr[d];
      attw[t * 9 + rem] = acc * 0.125f;                    // 1/sqrt(64)
    }
    __syncthreads();
    if (tid < 96) {                                        // softmax over sk
      float* p = &attw[(tid / 3) * 9 + (tid % 3) * 3];
      float mx = fmaxf(p[0], fmaxf(p[1], p[2]));
      float e0 = __expf(p[0] - mx), e1 = __expf(p[1] - mx), e2 = __expf(p[2] - mx);
      float inv = 1.f / (e0 + e1 + e2);
      p[0] = e0 * inv; p[1] = e1 * inv; p[2] = e2 * inv;
    }
    __syncthreads();
    for (int i = tid; i < 96 * 64; i += 768) {             // o = attn @ v  -> ost f16
      int row = i >> 6, d = i & 63;
      int t = row & 31;
      const float* aw = &attw[t * 9 + (row >> 5) * 3];
      float acc = aw[0] * (float)qkv[t * 200 + 128 + d]
                + aw[1] * (float)qkv[(32 + t) * 200 + 128 + d]
                + aw[2] * (float)qkv[(64 + t) * 200 + 128 + d];
      ost[row * 72 + d] = (_Float16)acc;
    }
    // ---- out-proj accumulate: Cop += o_h @ W_out[:, h*64:(h+1)*64]^T
    #pragma unroll
    for (int ks = 0; ks < 2; ++ks) {
      __syncthreads();                                     // qkv dead, o written
      if (wave == 0) {
        tdm_load_2d(ldsoff(Wst), wouth + h * DHEAD + ks * 32, 512, 32, DMODEL, 3, 3);
        __builtin_amdgcn_s_wait_tensorcnt(0);
      }
      __syncthreads();
      hvec16 a = ldA(ost, 72, mt * 16 + ln, ks * 32, hf);
      #pragma unroll
      for (int j = 0; j < 8; ++j) {
        hvec16 b = ldB(Wst, 40, ngp * 128 + j * 16 + ln, 0, hf);
        Cop[j] = wmma16(a, b, Cop[j]);
      }
    }
  }

  // ---- epilogue: residual + b_out, stream-gate softmax, gated write (f16)
  const int Nb = ngp * 128;
  #pragma unroll
  for (int j = 0; j < 8; ++j) {
    int n = Nb + j * 16 + ln;
    float bo = b_out[n];
    #pragma unroll
    for (int r = 0; r < 8; ++r) {
      int m = mt * 16 + r + 8 * hf;
      const float* xp = pick3(xk, xc, xs, m >> 5);
      Cop[j][r] += bo + xp[(size_t)(b0 + (m & 31)) * DMODEL + n];
    }
  }
  __syncthreads();
  if (tid < 96) lgts[tid] = b_gate[tid % 3];
  __syncthreads();
  #pragma unroll
  for (int r = 0; r < 8; ++r) {
    int m = mt * 16 + r + 8 * hf;
    int s = m >> 5, t = m & 31;
    float p0 = 0.f, p1 = 0.f, p2 = 0.f;
    #pragma unroll
    for (int j = 0; j < 8; ++j) {
      int f = s * DMODEL + Nb + j * 16 + ln;
      float v = Cop[j][r];
      p0 += v * w_gate[f];
      p1 += v * w_gate[D3 + f];
      p2 += v * w_gate[2 * D3 + f];
    }
    atomicAdd(&lgts[t * 3 + 0], p0);
    atomicAdd(&lgts[t * 3 + 1], p1);
    atomicAdd(&lgts[t * 3 + 2], p2);
  }
  __syncthreads();
  if (tid < 32) {
    float a0 = lgts[tid * 3], a1 = lgts[tid * 3 + 1], a2 = lgts[tid * 3 + 2];
    float mx = fmaxf(a0, fmaxf(a1, a2));
    float e0 = __expf(a0 - mx), e1 = __expf(a1 - mx), e2 = __expf(a2 - mx);
    float inv = 1.f / (e0 + e1 + e2);
    gts[tid * 3 + 0] = e0 * inv; gts[tid * 3 + 1] = e1 * inv; gts[tid * 3 + 2] = e2 * inv;
  }
  __syncthreads();
  #pragma unroll
  for (int r = 0; r < 8; ++r) {
    int m = mt * 16 + r + 8 * hf;
    int s = m >> 5, t = m & 31;
    float g = gts[t * 3 + s];
    #pragma unroll
    for (int j = 0; j < 8; ++j) {
      int n = Nb + j * 16 + ln;
      gated[(size_t)(b0 + t) * D3 + s * DMODEL + n] = (_Float16)(Cop[j][r] * g);
    }
  }
}

// ---------------------------------------------------------------- K2a: h = silu(gated @ w1^T + b1) -> f16
// Double-buffered TDM pipeline: chunk i+1 streams to LDS while chunk i is consumed by WMMA.
__global__ __launch_bounds__(512) void k2a_mlp1(
    const _Float16* __restrict__ gated, const _Float16* __restrict__ w1h,
    const float* __restrict__ b1, _Float16* __restrict__ hbuf)
{
  __shared__ __align__(16) _Float16 sB[2][512 * 40];
  __shared__ __align__(16) _Float16 sA[2][64 * 40];
  const int tid = threadIdx.x, wave = tid >> 5, lane = tid & 31;
  const int ln = lane & 15, hf = lane >> 4;
  const int mt = wave & 3, ngp = wave >> 2;
  const int b0 = blockIdx.x * 64;

  for (int nh = 0; nh < 2; ++nh) {
    fvec8 C[8];
    #pragma unroll
    for (int j = 0; j < 8; ++j)
      #pragma unroll
      for (int r = 0; r < 8; ++r) C[j][r] = 0.f;

    __syncthreads();                      // previous consumers of all buffers done
    if (wave == 0) {
      tdm_load_2d(ldsoff(sB[0]), w1h + (size_t)(nh * 512) * D3, 512, 32, D3, 3, 3);
      tdm_load_2d(ldsoff(sA[0]), gated + (size_t)b0 * D3, 64, 32, D3, 3, 3);
    }
    for (int ic = 0; ic < 48; ++ic) {
      __syncthreads();                    // consumers of buf[(ic+1)&1] (iter ic-1) done
      if (wave == 0) {
        if (ic + 1 < 48) {
          int kn = (ic + 1) * 32;
          tdm_load_2d(ldsoff(sB[(ic + 1) & 1]), w1h + (size_t)(nh * 512) * D3 + kn, 512, 32, D3, 3, 3);
          tdm_load_2d(ldsoff(sA[(ic + 1) & 1]), gated + (size_t)b0 * D3 + kn, 64, 32, D3, 3, 3);
          __builtin_amdgcn_s_wait_tensorcnt(2);   // chunk ic resident; chunk ic+1 in flight
        } else {
          __builtin_amdgcn_s_wait_tensorcnt(0);
        }
      }
      __syncthreads();
      const _Float16* bufA = sA[ic & 1];
      const _Float16* bufB = sB[ic & 1];
      hvec16 a = ldA(bufA, 40, mt * 16 + ln, 0, hf);
      #pragma unroll
      for (int j = 0; j < 8; ++j) {
        hvec16 b = ldB(bufB, 40, ngp * 128 + j * 16 + ln, 0, hf);
        C[j] = wmma16(a, b, C[j]);
      }
    }
    #pragma unroll
    for (int j = 0; j < 8; ++j) {
      int n = nh * 512 + ngp * 128 + j * 16 + ln;
      float bb = b1[n];
      #pragma unroll
      for (int r = 0; r < 8; ++r) {
        int m = mt * 16 + r + 8 * hf;
        float v = C[j][r] + bb;
        v = v / (1.f + __expf(-v));                         // SiLU
        hbuf[(size_t)(b0 + m) * D2 + n] = (_Float16)v;
      }
    }
  }
}

// ---------------------------------------------------------------- K2b: out = LN(h @ w2^T + b2) -> fp32
__global__ __launch_bounds__(512) void k2b_mlp2_ln(
    const _Float16* __restrict__ hbuf, const _Float16* __restrict__ w2h,
    const float* __restrict__ b2, const float* __restrict__ gamma,
    const float* __restrict__ beta, float* __restrict__ out)
{
  __shared__ __align__(16) _Float16 sB[2][512 * 40];
  __shared__ __align__(16) _Float16 sA[2][64 * 40];
  __shared__ float rsum[64], rsq[64], muv[64], rsg[64];
  const int tid = threadIdx.x, wave = tid >> 5, lane = tid & 31;
  const int ln = lane & 15, hf = lane >> 4;
  const int mt = wave & 3, ngp = wave >> 2;
  const int b0 = blockIdx.x * 64;

  fvec8 C[8];
  #pragma unroll
  for (int j = 0; j < 8; ++j)
    #pragma unroll
    for (int r = 0; r < 8; ++r) C[j][r] = 0.f;

  if (wave == 0) {
    tdm_load_2d(ldsoff(sB[0]), w2h, 512, 32, D2, 3, 3);
    tdm_load_2d(ldsoff(sA[0]), hbuf + (size_t)b0 * D2, 64, 32, D2, 3, 3);
  }
  for (int ic = 0; ic < 32; ++ic) {
    __syncthreads();
    if (wave == 0) {
      if (ic + 1 < 32) {
        int kn = (ic + 1) * 32;
        tdm_load_2d(ldsoff(sB[(ic + 1) & 1]), w2h + kn, 512, 32, D2, 3, 3);
        tdm_load_2d(ldsoff(sA[(ic + 1) & 1]), hbuf + (size_t)b0 * D2 + kn, 64, 32, D2, 3, 3);
        __builtin_amdgcn_s_wait_tensorcnt(2);
      } else {
        __builtin_amdgcn_s_wait_tensorcnt(0);
      }
    }
    __syncthreads();
    const _Float16* bufA = sA[ic & 1];
    const _Float16* bufB = sB[ic & 1];
    hvec16 a = ldA(bufA, 40, mt * 16 + ln, 0, hf);
    #pragma unroll
    for (int j = 0; j < 8; ++j) {
      hvec16 b = ldB(bufB, 40, ngp * 128 + j * 16 + ln, 0, hf);
      C[j] = wmma16(a, b, C[j]);
    }
  }
  if (tid < 64) { rsum[tid] = 0.f; rsq[tid] = 0.f; }
  __syncthreads();
  #pragma unroll
  for (int r = 0; r < 8; ++r) {
    int m = mt * 16 + r + 8 * hf;
    float ps = 0.f, pss = 0.f;
    #pragma unroll
    for (int j = 0; j < 8; ++j) {
      int n = ngp * 128 + j * 16 + ln;
      float v = C[j][r] + b2[n];
      C[j][r] = v;
      ps += v; pss += v * v;
    }
    atomicAdd(&rsum[m], ps);
    atomicAdd(&rsq[m], pss);
  }
  __syncthreads();
  if (tid < 64) {
    float mu = rsum[tid] * (1.f / 512.f);
    float var = rsq[tid] * (1.f / 512.f) - mu * mu;
    muv[tid] = mu;
    rsg[tid] = rsqrtf(var + 1e-5f);
  }
  __syncthreads();
  #pragma unroll
  for (int r = 0; r < 8; ++r) {
    int m = mt * 16 + r + 8 * hf;
    float mu = muv[m], rv = rsg[m];
    #pragma unroll
    for (int j = 0; j < 8; ++j) {
      int n = ngp * 128 + j * 16 + ln;
      out[(size_t)(b0 + m) * DMODEL + n] = (C[j][r] - mu) * rv * gamma[n] + beta[n];
    }
  }
}

// ---------------------------------------------------------------- launch
extern "C" void kernel_launch(void* const* d_in, const int* in_sizes, int n_in,
                              void* d_out, int out_size, void* d_ws, size_t ws_size,
                              hipStream_t stream) {
  (void)in_sizes; (void)n_in; (void)out_size; (void)ws_size;
  const float* xk     = (const float*)d_in[0];
  const float* xc     = (const float*)d_in[1];
  const float* xsp    = (const float*)d_in[2];
  const float* w_in   = (const float*)d_in[3];
  const float* b_in   = (const float*)d_in[4];
  const float* w_out  = (const float*)d_in[5];
  const float* b_out  = (const float*)d_in[6];
  const float* w_gate = (const float*)d_in[7];
  const float* b_gate = (const float*)d_in[8];
  const float* w1     = (const float*)d_in[9];
  const float* b1     = (const float*)d_in[10];
  const float* w2     = (const float*)d_in[11];
  const float* b2     = (const float*)d_in[12];
  const float* gamma  = (const float*)d_in[13];
  const float* beta   = (const float*)d_in[14];

  unsigned char* ws = (unsigned char*)d_ws;
  _Float16* winh  = (_Float16*)ws;  ws += (size_t)D3 * DMODEL * 2;
  _Float16* wouth = (_Float16*)ws;  ws += (size_t)DMODEL * DMODEL * 2;
  _Float16* w1h   = (_Float16*)ws;  ws += (size_t)D2 * D3 * 2;
  _Float16* w2h   = (_Float16*)ws;  ws += (size_t)DMODEL * D2 * 2;
  _Float16* gated = (_Float16*)ws;  ws += (size_t)B_TOT * D3 * 2;
  _Float16* hbuf  = (_Float16*)ws;  ws += (size_t)B_TOT * D2 * 2;

  k0_f32_to_f16<<<(D3 * DMODEL + 255) / 256, 256, 0, stream>>>(w_in, winh, D3 * DMODEL);
  k0_f32_to_f16<<<(DMODEL * DMODEL + 255) / 256, 256, 0, stream>>>(w_out, wouth, DMODEL * DMODEL);
  k0_f32_to_f16<<<(D2 * D3 + 255) / 256, 256, 0, stream>>>(w1, w1h, D2 * D3);
  k0_f32_to_f16<<<(DMODEL * D2 + 255) / 256, 256, 0, stream>>>(w2, w2h, DMODEL * D2);

  k1_attn_gate<<<B_TOT / 32, 768, 0, stream>>>(xk, xc, xsp, winh, b_in, wouth, b_out,
                                               w_gate, b_gate, gated);
  k2a_mlp1<<<B_TOT / 64, 512, 0, stream>>>(gated, w1h, b1, hbuf);
  k2b_mlp2_ln<<<B_TOT / 64, 512, 0, stream>>>(hbuf, w2h, b2, gamma, beta, (float*)d_out);
}
